// Net_50354196578736
// MI455X (gfx1250) — compile-verified, matrix operand order checked by reading, not run on hardware
//
#include <hip/hip_runtime.h>
#include <hip/hip_bf16.h>

// ---------------------------------------------------------------------------
// CDNA5 (gfx1250) WMMA types & fragment index helper
// ---------------------------------------------------------------------------
typedef _Float16 half_t;
typedef _Float16 v16h __attribute__((ext_vector_type(16)));
typedef _Float16 v8h  __attribute__((ext_vector_type(8)));
typedef float    v8f  __attribute__((ext_vector_type(8)));

// 16-bit A/B matrix VGPR striping (ISA 7.12.2):
//   element e = 2*v + p of the per-lane v16h maps to
//   k = 2*(v&3) + p + 8*(lane>=16) + 16*(v>=4)
// => lane's 16 elements are two contiguous aligned 8-runs:
//    e 0..7  -> k = base+0 .. base+7
//    e 8..15 -> k = base+16 .. base+23      (base = kt*32 + 8*(lane>=16))
__device__ __forceinline__ int frag_k(int e, int kh) {
    int v = e >> 1, p = e & 1;
    return ((v & 3) << 1) + p + (kh << 3) + ((v >> 2) << 4);
}

union BFrag { v16h v; uint4 q[2]; };

// ---------------------------------------------------------------------------
// One-time weight packers -> f16, WMMA A-fragment striping.
// Packed layout: dst[((mtG*KT + kt)*32 + lane)*16 + e]  with
//   row = mtG*16 + (lane&15),  k = kt*32 + frag_k(e, lane>>4)
// ---------------------------------------------------------------------------

// Plain GEMM weights (conv3/conv4/conv5): k maps 1:1 to source k.
__global__ void pack_wmma_a_kernel(const float* __restrict__ src, // (Mreal,Kreal)
                                   half_t* __restrict__ dst,
                                   int Mreal, int Kreal, int KT, int n)
{
    int id = blockIdx.x * blockDim.x + threadIdx.x;
    if (id >= n) return;
    int e    = id & 15;
    int lane = (id >> 4) & 31;
    int kt   = (id >> 9) % KT;
    int mtG  = (id >> 9) / KT;
    int row  = mtG * 16 + (lane & 15);
    int k    = kt * 32 + frag_k(e, lane >> 4);
    float v  = (row < Mreal && k < Kreal) ? src[row * Kreal + k] : 0.f;
    dst[id] = (half_t)v;
}

// ConvLSTM weights with tap-major permuted K:
//   kp = seg*CTp + cp,  seg = ky*3+kx (tap),
//   cp < F        -> h channel   (orig cin = CIN + cp)
//   F <= cp < F+CIN -> x channel (orig cin = cp - F)
//   else zero pad
// orig k = cin*9 + seg   (weight layout (4F, CIN+F, 3, 3))
__global__ void pack_clstm_w_kernel(const float* __restrict__ src,
                                    half_t* __restrict__ dst,
                                    int F, int CIN, int CTp, int KT, int n)
{
    int id = blockIdx.x * blockDim.x + threadIdx.x;
    if (id >= n) return;
    int e    = id & 15;
    int lane = (id >> 4) & 31;
    int kt   = (id >> 9) % KT;
    int mtG  = (id >> 9) / KT;
    int row  = mtG * 16 + (lane & 15);
    int kp   = kt * 32 + frag_k(e, lane >> 4);
    float v  = 0.f;
    if (kp < 9 * CTp) {
        int seg = kp / CTp, cp = kp % CTp;
        int cin = -1;
        if (cp < F)            cin = CIN + cp;
        else if (cp < F + CIN) cin = cp - F;
        if (cin >= 0) v = src[(row * (CIN + F) + cin) * 9 + seg];
    }
    dst[id] = (half_t)v;
}

__global__ void f32_to_f16_kernel(const float* __restrict__ src,
                                  half_t* __restrict__ dst, int n)
{
    int i = blockIdx.x * blockDim.x + threadIdx.x;
    if (i < n) dst[i] = (half_t)src[i];
}

// ---------------------------------------------------------------------------
// Per-step channel fills into the NHWC concat buffer xcat[pos][CTp].
// ---------------------------------------------------------------------------
// layer1: x_t (B,1,32,8,256) f16 -> channel 32 of xcat (CTp=40)
__global__ void fill_x1_kernel(const half_t* __restrict__ xh,
                               half_t* __restrict__ xcat, int t)
{
    int i = blockIdx.x * blockDim.x + threadIdx.x;
    if (i >= 8 * 2048) return;                 // B * H*W
    int bb = i >> 11, rem = i & 2047;
    xcat[i * 40 + 32] = xh[(bb * 32 + t) * 2048 + rem];
}

// layer2: act1p slice t, NHWC (B,16,4,128,32) -> channels 48..79 of xcat (CTp=80)
__global__ void fill_x2_kernel(const half_t* __restrict__ act1p,
                               half_t* __restrict__ xcat, int t)
{
    int i = blockIdx.x * blockDim.x + threadIdx.x;
    if (i >= 8 * 512 * 32) return;
    int c = i & 31, p = i >> 5;                // p = (bb*4+y)*128+x
    int bb = p >> 9, rem = p & 511;
    xcat[p * 80 + 48 + c] = act1p[((bb * 16 + t) * 512 + rem) * 32 + c];
}

// ---------------------------------------------------------------------------
// ConvLSTM single-timestep kernel.
//   gates z = Wp * im2col(xcat)   K' = 9*CTp (tap-major), N = B*H*W
// Block = 4 waves, each wave owns one 16-position N-tile and ALL 4F gate
// rows (NTILES = 4*F/16 accumulators), so each im2col B fragment (2 aligned
// b128 loads) feeds NTILES WMMAs.  The full packed A k-slice (NTILES*1KB)
// is staged through LDS per k-step and shared by the 4 waves.
// h is written as f16 straight into the next step's xcat channel block
// (b128 stores); c stays f32, NHWC.
// ---------------------------------------------------------------------------
template<int CIN, int F, int H, int W>
__global__ void __launch_bounds__(128)
clstm_step_kernel(const half_t* __restrict__ xcatR, // (B*H*W, CTp) [h|x|pad]
                  const float*  __restrict__ cprev, // (B*H*W, F)
                  const half_t* __restrict__ wp,    // packed A
                  const float*  __restrict__ bias,  // (4F)
                  half_t* __restrict__ xcatW,       // h -> channels 0..F-1
                  float*  __restrict__ cnext)       // (B*H*W, F)
{
    constexpr int CTp    = ((CIN + F + 7) / 8) * 8;
    constexpr int Kp     = 9 * CTp;
    constexpr int KT     = (Kp + 31) / 32;
    constexpr int FT     = F / 16;
    constexpr int NTILES = 4 * FT;                 // i,f,g,o x FT
    constexpr int HW     = H * W;

    __shared__ __align__(32) half_t aS[NTILES * 32 * 16];

    const int lane = threadIdx.x & 31;
    const int wv   = threadIdx.x >> 5;
    const int nt   = blockIdx.x * 4 + wv;

    const int nl  = lane & 15;
    const int kh  = lane >> 4;
    const int pos = nt * 16 + nl;
    const int bb  = pos / HW;
    const int yy  = (pos % HW) / W;
    const int xx  = pos % W;

    v8f acc[NTILES] = {};

    for (int kt = 0; kt < KT; ++kt) {
        // --- cooperative LDS stage of all NTILES packed A fragments ---
        {
            const uint4* ws4 = (const uint4*)wp;
            uint4* as4 = (uint4*)aS;
            #pragma unroll
            for (int j = 0; j < NTILES / 2; ++j) {
                int q   = threadIdx.x + j * 128;   // 0 .. NTILES*64-1
                int mtG = q >> 6, qi = q & 63;
                as4[mtG * 64 + qi] = ws4[(mtG * KT + kt) * 64 + qi];
            }
        }
        __syncthreads();

        if (threadIdx.x == 0 && kt + 1 < KT)       // global_prefetch_b8
            __builtin_prefetch(&wp[(kt + 1) * 512], 0, 1);

        // --- B fragment: two aligned b128 loads from NHWC concat buffer ---
        BFrag bu;
        #pragma unroll
        for (int ri = 0; ri < 2; ++ri) {
            int rb = kt * 32 + kh * 8 + ri * 16;   // aligned 8-run base
            uint4 d = {0u, 0u, 0u, 0u};
            if (rb < Kp) {
                int seg = rb / CTp, off = rb % CTp;
                int y2 = yy + seg / 3 - 1;
                int x2 = xx + seg % 3 - 1;
                if (y2 >= 0 && y2 < H && x2 >= 0 && x2 < W)
                    d = *(const uint4*)&xcatR[((bb * H + y2) * W + x2) * CTp + off];
            }
            bu.q[ri] = d;
        }

        // --- NTILES WMMAs sharing bu.v; A via ds b128 reads ---
        #pragma unroll
        for (int mtG = 0; mtG < NTILES; ++mtG) {
            v16h a = *(const v16h*)&aS[(mtG * 32 + lane) * 16];
            acc[mtG] = __builtin_amdgcn_wmma_f32_16x16x32_f16(
                false, a, false, bu.v, (short)0, acc[mtG], false, false);
        }
        __syncthreads();
    }

    // --- fused LSTM pointwise.  acc[g*FT+m]: row = m*16 + 8*kh + r ---
    #pragma unroll
    for (int m = 0; m < FT; ++m) {
        const int fcBase = m * 16 + kh * 8;
        const float* cP = &cprev[pos * F + fcBase];
        float*       cN = &cnext[pos * F + fcBase];
        v8h hv;
        #pragma unroll
        for (int r = 0; r < 8; ++r) {
            int fc   = fcBase + r;
            float zi = acc[0 * FT + m][r] + bias[0 * F + fc];
            float zf = acc[1 * FT + m][r] + bias[1 * F + fc];
            float zg = acc[2 * FT + m][r] + bias[2 * F + fc];
            float zo = acc[3 * FT + m][r] + bias[3 * F + fc];
            float co = cP[r];
            float si = 1.f / (1.f + __expf(-zi));
            float sf = 1.f / (1.f + __expf(-zf));
            float so = 1.f / (1.f + __expf(-zo));
            float cn = sf * co + si * tanhf(zg);
            float hn = so * tanhf(cn);
            cN[r] = cn;
            hv[r] = (half_t)hn;
        }
        *(v8h*)&xcatW[pos * CTp + fcBase] = hv;    // b128 h store
    }
}

// ---------------------------------------------------------------------------
// Streamed MaxPool3d(2,2,2): layer1 variant, h from xcat (ch 0..31),
// output NHWC act1p (B,16,4,128,32).
// ---------------------------------------------------------------------------
__global__ void pool1_kernel(const half_t* __restrict__ hEven,  // xcat bufs
                             const half_t* __restrict__ hOdd,
                             half_t* __restrict__ out, int tp)
{
    int i = blockIdx.x * blockDim.x + threadIdx.x;
    if (i >= 8 * 4 * 128 * 32) return;
    int c  = i & 31, p = i >> 5;
    int xo = p % 128, yo = (p / 128) % 4, bb = p / 512;
    auto at = [&](const half_t* b, int y, int x) {
        return (float)b[((bb * 8 + y) * 256 + x) * 40 + c];
    };
    int y = 2 * yo, x = 2 * xo;
    float m = at(hEven, y, x);
    m = fmaxf(m, at(hEven, y, x + 1));
    m = fmaxf(m, at(hEven, y + 1, x));
    m = fmaxf(m, at(hEven, y + 1, x + 1));
    m = fmaxf(m, at(hOdd, y, x));
    m = fmaxf(m, at(hOdd, y, x + 1));
    m = fmaxf(m, at(hOdd, y + 1, x));
    m = fmaxf(m, at(hOdd, y + 1, x + 1));
    out[((bb * 16 + tp) * 512 + (yo * 128 + xo)) * 32 + c] = (half_t)m;
}

// layer2 variant: h from xcat2 (ch 0..47), output NCHW act2p (8,48,8,2,64)
// (row-of-64 contiguous so conv3 B-runs are aligned b128s).
__global__ void pool2_kernel(const half_t* __restrict__ hEven,
                             const half_t* __restrict__ hOdd,
                             half_t* __restrict__ out, int tp)
{
    int i = blockIdx.x * blockDim.x + threadIdx.x;
    if (i >= 8 * 48 * 2 * 64) return;
    int xo = i % 64, yo = (i / 64) % 2, f = (i / 128) % 48, bb = i / (128 * 48);
    auto at = [&](const half_t* b, int y, int x) {
        return (float)b[((bb * 4 + y) * 128 + x) * 80 + f];
    };
    int y = 2 * yo, x = 2 * xo;
    float m = at(hEven, y, x);
    m = fmaxf(m, at(hEven, y, x + 1));
    m = fmaxf(m, at(hEven, y + 1, x));
    m = fmaxf(m, at(hEven, y + 1, x + 1));
    m = fmaxf(m, at(hOdd, y, x));
    m = fmaxf(m, at(hOdd, y, x + 1));
    m = fmaxf(m, at(hOdd, y + 1, x));
    m = fmaxf(m, at(hOdd, y + 1, x + 1));
    out[(((bb * 48 + f) * 8 + tp) * 2 + yo) * 64 + xo] = (half_t)m;
}

// ---------------------------------------------------------------------------
// conv3: (B,48,16,64) f16 --(256,9216)--> ELU -> act3t (B,14,256) pos-major.
// B fragments: aligned b128 runs inside (cin,ky) rows of 64.
// ---------------------------------------------------------------------------
__global__ void __launch_bounds__(32)
conv3_kernel(const half_t* __restrict__ in,   // (B,48,16,64)
             const half_t* __restrict__ wp,   // packed (16 x 288)
             const float*  __restrict__ bias,
             half_t* __restrict__ out)        // (B*14, 256) pos-major
{
    constexpr int KT = 9216 / 32;
    const int lane = threadIdx.x & 31;
    const int mt = blockIdx.x % 16;
    const int nt = blockIdx.x / 16;
    const int nl = lane & 15, kh = lane >> 4;
    const int pos = nt * 16 + nl;            // 0..111
    const int bb = pos / 14;
    const int oy = pos % 14;

    v8f acc = {};
    for (int kt = 0; kt < KT; ++kt) {
        v16h a = *(const v16h*)&wp[((mt * KT + kt) * 32 + lane) * 16];
        BFrag bu;
        #pragma unroll
        for (int ri = 0; ri < 2; ++ri) {
            int rb  = kt * 32 + kh * 8 + ri * 16;
            int cin = rb / 192;
            int rem = rb % 192;
            int ky  = rem >> 6, kx = rem & 63;
            bu.q[ri] = *(const uint4*)&in[((bb * 48 + cin) * 16 + oy + ky) * 64 + kx];
        }
        acc = __builtin_amdgcn_wmma_f32_16x16x32_f16(
            false, a, false, bu.v, (short)0, acc, false, false);
    }
    v8h hv;
    #pragma unroll
    for (int r = 0; r < 8; ++r) {
        int m = mt * 16 + kh * 8 + r;
        float v = acc[r] + bias[m];
        hv[r] = (half_t)(v > 0.f ? v : (__expf(v) - 1.f));   // ELU
    }
    *(v8h*)&out[pos * 256 + mt * 16 + kh * 8] = hv;          // b128 store
}

// ---------------------------------------------------------------------------
// 1x1 conv as WMMA GEMM, pos-major f16 B (contiguous K runs -> b128 loads).
// conv4: f16 pos-major out + ELU.  conv5: guarded f32 NCHW out.
// ---------------------------------------------------------------------------
template<int M, int K, int MR, bool ELU, bool POSMAJOR_OUT>
__global__ void __launch_bounds__(32)
gemm1x1_kernel(const half_t* __restrict__ in,   // (B*14, K) pos-major
               const half_t* __restrict__ wp,   // packed (M/16 x K/32)
               const float*  __restrict__ bias, // (MR)
               void* __restrict__ outv)
{
    constexpr int KT = K / 32;
    const int lane = threadIdx.x & 31;
    const int mt = blockIdx.x % (M / 16);
    const int nt = blockIdx.x / (M / 16);
    const int nl = lane & 15, kh = lane >> 4;
    const int pos = nt * 16 + nl;
    const int bb = pos / 14;
    const int oy = pos % 14;

    v8f acc = {};
    for (int kt = 0; kt < KT; ++kt) {
        v16h a = *(const v16h*)&wp[((mt * KT + kt) * 32 + lane) * 16];
        BFrag bu;
        bu.q[0] = *(const uint4*)&in[pos * K + kt * 32 + kh * 8];
        bu.q[1] = *(const uint4*)&in[pos * K + kt * 32 + kh * 8 + 16];
        acc = __builtin_amdgcn_wmma_f32_16x16x32_f16(
            false, a, false, bu.v, (short)0, acc, false, false);
    }
    if (POSMAJOR_OUT) {
        half_t* out = (half_t*)outv;
        v8h hv;
        #pragma unroll
        for (int r = 0; r < 8; ++r) {
            int m = mt * 16 + kh * 8 + r;
            float v = acc[r] + bias[m];
            if (ELU) v = v > 0.f ? v : (__expf(v) - 1.f);
            hv[r] = (half_t)v;
        }
        *(v8h*)&out[pos * MR + mt * 16 + kh * 8] = hv;
    } else {
        float* out = (float*)outv;
        #pragma unroll
        for (int r = 0; r < 8; ++r) {
            int m = mt * 16 + kh * 8 + r;
            if (m < MR) {
                float v = acc[r] + bias[m];
                if (ELU) v = v > 0.f ? v : (__expf(v) - 1.f);
                out[(bb * MR + m) * 14 + oy] = v;
            }
        }
    }
}

// ---------------------------------------------------------------------------
// Host orchestration (all on `stream`, graph-capture safe)
// ---------------------------------------------------------------------------
extern "C" void kernel_launch(void* const* d_in, const int* in_sizes, int n_in,
                              void* d_out, int out_size, void* d_ws, size_t ws_size,
                              hipStream_t stream) {
    const float* x  = (const float*)d_in[0];
    const float* w1 = (const float*)d_in[1];
    const float* b1 = (const float*)d_in[2];
    const float* w2 = (const float*)d_in[3];
    const float* b2 = (const float*)d_in[4];
    const float* w3 = (const float*)d_in[5];
    const float* b3 = (const float*)d_in[6];
    const float* w4 = (const float*)d_in[7];
    const float* b4 = (const float*)d_in[8];
    const float* w5 = (const float*)d_in[9];
    const float* b5 = (const float*)d_in[10];

    // workspace carve-up (256B aligned)
    size_t off = 0;
    auto carveB = [&](size_t bytes) {
        void* p = (char*)d_ws + off;
        off += (bytes + 255) & ~(size_t)255;
        return p;
    };
    auto carveF = [&](size_t n) { return (float*) carveB(n * 4); };
    auto carveH = [&](size_t n) { return (half_t*)carveB(n * 2); };

    float*  c1[2]   = { carveF(524288), carveF(524288) };   // (16384, 32) f32
    float*  c2[2]   = { carveF(196608), carveF(196608) };   // (4096, 48)  f32
    half_t* xcat1[2]= { carveH(16384 * 40), carveH(16384 * 40) }; // [h32|x1|pad7]
    half_t* xcat2[2]= { carveH(4096  * 80), carveH(4096  * 80) }; // [h48|x32]
    half_t* xh      = carveH(524288);                        // x f16
    half_t* act1p   = carveH(8 * 16 * 512 * 32);             // NHWC (B,16,4,128,32)
    half_t* act2p   = carveH(8 * 48 * 8 * 2 * 64);           // NCHW (B,48,8,2,64)
    half_t* act3t   = carveH(112 * 256);                     // pos-major
    half_t* act4t   = carveH(112 * 128);                     // pos-major
    half_t* w1p     = carveH(8  * 12  * 512);                // packed weights
    half_t* w2p     = carveH(12 * 23  * 512);
    half_t* w3p     = carveH(16 * 288 * 512);
    half_t* w4p     = carveH(8  * 8   * 512);
    half_t* w5p     = carveH(6  * 4   * 512);

    // zero h0 / channel pads / c0
    hipMemsetAsync(xcat1[0], 0, 16384 * 40 * sizeof(half_t), stream);
    hipMemsetAsync(xcat1[1], 0, 16384 * 40 * sizeof(half_t), stream);
    hipMemsetAsync(xcat2[0], 0, 4096 * 80 * sizeof(half_t), stream);
    hipMemsetAsync(xcat2[1], 0, 4096 * 80 * sizeof(half_t), stream);
    hipMemsetAsync(c1[0], 0, 524288 * sizeof(float), stream);
    hipMemsetAsync(c2[0], 0, 196608 * sizeof(float), stream);

    // ---- one-time f16 conversion + weight packing ----
    auto grid = [](int n) { return (n + 255) / 256; };
    f32_to_f16_kernel<<<grid(524288), 256, 0, stream>>>(x, xh, 524288);
    pack_clstm_w_kernel<<<grid(49152),  256, 0, stream>>>(w1, w1p, 32, 1,  40, 12, 49152);
    pack_clstm_w_kernel<<<grid(141312), 256, 0, stream>>>(w2, w2p, 48, 32, 80, 23, 141312);
    pack_wmma_a_kernel<<<grid(2359296), 256, 0, stream>>>(w3, w3p, 256, 9216, 288, 2359296);
    pack_wmma_a_kernel<<<grid(32768),   256, 0, stream>>>(w4, w4p, 128, 256,  8,   32768);
    pack_wmma_a_kernel<<<grid(12288),   256, 0, stream>>>(w5, w5p, 88,  128,  4,   12288);

    // ---- ConvLSTM layer 1 (T=32), streamed pool.  1024 N-tiles / 4 = 256 blocks
    for (int t = 0; t < 32; ++t) {
        fill_x1_kernel<<<grid(16384), 256, 0, stream>>>(xh, xcat1[t & 1], t);
        clstm_step_kernel<1, 32, 8, 256><<<256, 128, 0, stream>>>(
            xcat1[t & 1], c1[t & 1], w1p, b1, xcat1[(t + 1) & 1], c1[(t + 1) & 1]);
        if (t & 1)   // h_{t-1} in xcat1[1], h_t in xcat1[0]
            pool1_kernel<<<grid(131072), 256, 0, stream>>>(
                xcat1[1], xcat1[0], act1p, t >> 1);
    }

    // ---- ConvLSTM layer 2 (T=16), streamed pool.  256 N-tiles / 4 = 64 blocks
    for (int t = 0; t < 16; ++t) {
        fill_x2_kernel<<<grid(131072), 256, 0, stream>>>(act1p, xcat2[t & 1], t);
        clstm_step_kernel<32, 48, 4, 128><<<64, 128, 0, stream>>>(
            xcat2[t & 1], c2[t & 1], w2p, b2, xcat2[(t + 1) & 1], c2[(t + 1) & 1]);
        if (t & 1)
            pool2_kernel<<<grid(49152), 256, 0, stream>>>(
                xcat2[1], xcat2[0], act2p, t >> 1);
    }

    // ---- conv3 (K=9216) + ELU : 16 M-tiles x 7 N-tiles ----
    conv3_kernel<<<112, 32, 0, stream>>>(act2p, w3p, b3, act3t);

    // ---- conv4 (1x1, K=256) + ELU, pos-major f16 out ----
    gemm1x1_kernel<128, 256, 128, true, true><<<(128 / 16) * 7, 32, 0, stream>>>(
        act3t, w4p, b4, act4t);

    // ---- conv5 (1x1, K=128) -> d_out (8,88,14,1) f32 ----
    gemm1x1_kernel<96, 128, 88, false, false><<<(96 / 16) * 7, 32, 0, stream>>>(
        act4t, w5p, b5, d_out);
}
